// MultiHeadSelfAttention_36859409334987
// MI455X (gfx1250) — matrix operand order repaired and checked by reading, not compiled, not run on hardware
//
#include <hip/hip_runtime.h>
#include <hip/hip_bf16.h>
#include <math.h>

// ---------------------------------------------------------------------------
// MI455X (gfx1250) causal MHA + RoPE.  wave32, WMMA f32_16x16x32_f16,
// async global->LDS staging (ASYNCcnt), HW sin/cos for RoPE.
// ---------------------------------------------------------------------------

#define Bq 2
#define Sq 2048
#define Dm 1024
#define Hh 16
#define Dk 64

typedef _Float16 v16h __attribute__((ext_vector_type(16)));
typedef _Float16 v8h  __attribute__((ext_vector_type(8)));
typedef _Float16 v4h  __attribute__((ext_vector_type(4)));
typedef float    v8f  __attribute__((ext_vector_type(8)));
typedef float    v4f  __attribute__((ext_vector_type(4)));

__device__ __forceinline__ v16h cat8(v8h lo, v8h hi) {
  return __builtin_shufflevector(lo, hi, 0,1,2,3,4,5,6,7,8,9,10,11,12,13,14,15);
}

// A-fragment (16x32 f16, MxK): lane = row (l&15); lane-half selects the K split:
// halves at K = kc+hi*8+{0..7} and K = kc+16+hi*8+{0..7}.
__device__ __forceinline__ v16h load_a_frag(const _Float16* base, int kc, int hi) {
  v8h lo = *(const v8h*)(base + kc + hi * 8);
  v8h hh = *(const v8h*)(base + kc + 16 + hi * 8);
  return cat8(lo, hh);
}

// B-fragment (32x16 f16, KxN): lane = col (l&15); lane-half selects K = kc+hi*8+{0..7} and kc+16+hi*8+{0..7}.
// base = start of this column's K-run (storage [n][k], contiguous in K).
__device__ __forceinline__ v16h load_b_frag(const _Float16* base, int kc, int hi) {
  v8h lo = *(const v8h*)(base + kc + hi * 8);
  v8h hh = *(const v8h*)(base + kc + 16 + hi * 8);
  return cat8(lo, hh);
}

__device__ __forceinline__ v8f wmma16(v16h a, v16h b, v8f c) {
  return __builtin_amdgcn_wmma_f32_16x16x32_f16(false, a, false, b, (short)0, c,
                                                false, false);
}

__device__ __forceinline__ float redmax16(float v) {
  v = fmaxf(v, __shfl_xor(v, 1, 32));
  v = fmaxf(v, __shfl_xor(v, 2, 32));
  v = fmaxf(v, __shfl_xor(v, 4, 32));
  v = fmaxf(v, __shfl_xor(v, 8, 32));
  return v;
}
__device__ __forceinline__ float redsum16(float v) {
  v += __shfl_xor(v, 1, 32);
  v += __shfl_xor(v, 2, 32);
  v += __shfl_xor(v, 4, 32);
  v += __shfl_xor(v, 8, 32);
  return v;
}

// Async 32-byte global->LDS copy (two B128 ops, ASYNCcnt-tracked).
// LDS flat addressing uses addr[31:0] as the LDS byte offset (ISA 10.2).
__device__ __forceinline__ void async_copy32(const _Float16* gsrc, _Float16* lds) {
  unsigned long long ga = (unsigned long long)(size_t)gsrc;
  unsigned la = (unsigned)(size_t)lds;
  asm volatile("global_load_async_to_lds_b128 %0, %1, off"
               :: "v"(la), "v"(ga) : "memory");
  asm volatile("global_load_async_to_lds_b128 %0, %1, off offset:16"
               :: "v"(la), "v"(ga) : "memory");
}
__device__ __forceinline__ void async_wait0() {
  asm volatile("s_wait_asynccnt 0" ::: "memory");
}

// ---------------------------------------------------------------------------
// Kernel 1: Q/K/V = x @ W{q,k,v}, RoPE on Q,K, store f16 in [B,H,S,Dk].
// Block: 256 threads (8 waves), tile 64(M) x 32(N), K staged in 64-chunks.
// ---------------------------------------------------------------------------
__global__ __launch_bounds__(256) void qkv_rope_kernel(
    const float* __restrict__ x, const float* __restrict__ Wqp,
    const float* __restrict__ Wkp, const float* __restrict__ Wvp,
    const int* __restrict__ tp, _Float16* __restrict__ Qh,
    _Float16* __restrict__ Kh, _Float16* __restrict__ Vh) {
  __shared__ __align__(16) _Float16 Xt[64][72];   // [m][k]
  __shared__ __align__(16) _Float16 Wt[32][72];   // [n][k]

  const int tid  = threadIdx.x;
  const int wave = tid >> 5, lane = tid & 31;
  const int hi   = lane >> 4, ln = lane & 15;
  const int m0 = blockIdx.x * 64, n0 = blockIdx.y * 32;
  const int z  = blockIdx.z;
  const float* W = (z == 0) ? Wqp : (z == 1) ? Wkp : Wvp;
  _Float16* dst  = (z == 0) ? Qh : (z == 1) ? Kh : Vh;
  const int mw = (wave >> 1) * 16, nw = (wave & 1) * 16;

  v8f acc = {0.f, 0.f, 0.f, 0.f, 0.f, 0.f, 0.f, 0.f};

  for (int k0 = 0; k0 < Dm; k0 += 64) {
    __syncthreads();
    {  // x tile: 64 rows x 64 k, f32 -> f16
      int row = tid >> 2, seg = tid & 3;
      const float* src = x + (size_t)(m0 + row) * Dm + k0 + seg * 16;
      _Float16* d0 = &Xt[row][seg * 16];
#pragma unroll
      for (int j = 0; j < 4; ++j) {
        v4f f = *(const v4f*)(src + 4 * j);
        v4h hv = {(_Float16)f.x, (_Float16)f.y, (_Float16)f.z, (_Float16)f.w};
        *(v4h*)(d0 + 4 * j) = hv;
      }
    }
    {  // W tile: 64 k x 32 n, transposed into Wt[n][k], f32 -> f16
      int k = tid >> 3, ng = tid & 7;
#pragma unroll
      for (int i = 0; i < 2; ++i) {
        int kk = k + i * 32;
        v4f f = *(const v4f*)(W + (size_t)(k0 + kk) * Dm + n0 + ng * 4);
        Wt[ng * 4 + 0][kk] = (_Float16)f.x;
        Wt[ng * 4 + 1][kk] = (_Float16)f.y;
        Wt[ng * 4 + 2][kk] = (_Float16)f.z;
        Wt[ng * 4 + 3][kk] = (_Float16)f.w;
      }
    }
    __syncthreads();
    const _Float16* arow = &Xt[mw + ln][0];
    const _Float16* bcol = &Wt[nw + ln][0];
#pragma unroll
    for (int kc = 0; kc < 64; kc += 32) {
      acc = wmma16(load_a_frag(arow, kc, hi), load_b_frag(bcol, kc, hi), acc);
    }
  }

  // Epilogue: (optional RoPE) + scatter to [B,H,S,Dk] f16.
  const int n_g = n0 + nw + ln;
  const int hh = n_g >> 6, d = n_g & 63;
  // revolutions per unit position: 10000^(-(d&~1)/64) / (2*pi)
  const float freq_rev = exp2f(-(float)(d & ~1) * 0.20762050592f) * 0.15915494309f;
  const float sgn = (d & 1) ? 1.f : -1.f;
#pragma unroll
  for (int r = 0; r < 8; ++r) {
    int m_g = m0 + mw + 8 * hi + r;          // global token row
    int b = m_g >> 11, s = m_g & (Sq - 1);
    float val = acc[r];
    float y;
    if (z == 2) {
      y = val;                                // V: no RoPE
    } else {
      float partner = __shfl_xor(val, 1, 32); // even<->odd feature cols
      float t = (float)tp[m_g] * freq_rev;    // angle in revolutions
      float fr = t - rintf(t);                // [-0.5, 0.5] -> HW sin/cos domain
      float sn = __builtin_amdgcn_sinf(fr);   // sin(2*pi*fr) == sin(angle)
      float cs = __builtin_amdgcn_cosf(fr);
      y = val * cs + sgn * partner * sn;      // even: x1*c - x2*s ; odd: x1*s + x2*c
    }
    dst[(((size_t)b * Hh + hh) * Sq + s) * Dk + d] = (_Float16)y;
  }
}

// ---------------------------------------------------------------------------
// Kernel 2: flash attention per (b,h).  Block = 128 query rows (8 waves x 16).
// K tile staged via async global->LDS; V staged transposed; P re-laid via LDS.
// ---------------------------------------------------------------------------
__global__ __launch_bounds__(256) void attn_kernel(
    const _Float16* __restrict__ Qh, const _Float16* __restrict__ Kh,
    const _Float16* __restrict__ Vh, _Float16* __restrict__ Ah) {
  __shared__ __align__(16) _Float16 Kt[64][72];      // [key][d]
  __shared__ __align__(16) _Float16 Vt[64][72];      // [d][key]  (transposed)
  __shared__ __align__(16) _Float16 Pb[8][16][72];   // per-wave probs [qrow][key]

  const int tid  = threadIdx.x;
  const int wave = tid >> 5, lane = tid & 31;
  const int hi   = lane >> 4, ln = lane & 15;
  const int bh = blockIdx.y;            // b*16 + h
  const int b = bh >> 4, h = bh & 15;
  const int q0 = blockIdx.x * 128;
  const size_t base = (size_t)bh * Sq * Dk;
  const int qw = q0 + wave * 16;        // first query row of this wave

  // Preload Q fragments (rows qw..qw+15, K = d = 0..63)
  const _Float16* qrow = Qh + base + (size_t)(qw + ln) * Dk;
  v16h aq0 = load_a_frag(qrow, 0, hi);
  v16h aq1 = load_a_frag(qrow, 32, hi);

  float mr[8], lr[8], al[8];
  v8f zero = {0.f, 0.f, 0.f, 0.f, 0.f, 0.f, 0.f, 0.f};
  v8f o[4] = {zero, zero, zero, zero};
#pragma unroll
  for (int r = 0; r < 8; ++r) { mr[r] = -INFINITY; lr[r] = 0.f; }

  const int row = tid >> 2, seg = tid & 3;   // staging assignment
  const int nChunks = q0 / 64 + 2;           // causal: keys up to q0+127
  for (int c = 0; c < nChunks; ++c) {
    const int kb = c * 64;
    __syncthreads();
    // K tile: pure copy -> async DMA into LDS (ASYNCcnt)
    async_copy32(Kh + base + (size_t)(kb + row) * Dk + seg * 16,
                 &Kt[row][seg * 16]);
    {  // V tile: load + transpose-store into Vt[d][key]
      const _Float16* vs = Vh + base + (size_t)(kb + row) * Dk + seg * 16;
      v8h v0 = *(const v8h*)vs;
      v8h v1 = *(const v8h*)(vs + 8);
#pragma unroll
      for (int j = 0; j < 8; ++j) {
        Vt[seg * 16 + j][row]     = v0[j];
        Vt[seg * 16 + 8 + j][row] = v1[j];
      }
    }
    async_wait0();
    __syncthreads();

    // scores S = Q * K^T  (4 tiles of 16 keys, K-dim = 64)
    v8f sf[4];
#pragma unroll
    for (int ns = 0; ns < 4; ++ns) {
      const _Float16* kcol = &Kt[ns * 16 + ln][0];
      v8f sacc = zero;
      sacc = wmma16(aq0, load_b_frag(kcol, 0, hi), sacc);
      sacc = wmma16(aq1, load_b_frag(kcol, 32, hi), sacc);
      sf[ns] = sacc;
    }

    const bool needMask = (kb + 63) > qw;   // wave-uniform
#pragma unroll
    for (int r = 0; r < 8; ++r) {
      const int q_g = qw + 8 * hi + r;
      float v[4];
#pragma unroll
      for (int ns = 0; ns < 4; ++ns) {
        float s = sf[ns][r] * 0.125f;       // 1/sqrt(64)
        if (needMask) {
          int key = kb + ns * 16 + ln;
          if (key > q_g) s = -INFINITY;
        }
        v[ns] = s;
      }
      float rm = redmax16(fmaxf(fmaxf(v[0], v[1]), fmaxf(v[2], v[3])));
      float mnew = fmaxf(mr[r], rm);
      float a = expf(mr[r] - mnew);
      mr[r] = mnew;
      al[r] = a;
      float ps = 0.f;
#pragma unroll
      for (int ns = 0; ns < 4; ++ns) {
        float p = expf(v[ns] - mnew);
        ps += p;
        Pb[wave][8 * hi + r][ns * 16 + ln] = (_Float16)p;
      }
      lr[r] = lr[r] * a + redsum16(ps);
    }
    __syncthreads();

    // rescale running output, then O += P * V
#pragma unroll
    for (int ds = 0; ds < 4; ++ds)
#pragma unroll
      for (int r = 0; r < 8; ++r) o[ds][r] *= al[r];

    const _Float16* prow = &Pb[wave][ln][0];
    v16h ap0 = load_a_frag(prow, 0, hi);
    v16h ap1 = load_a_frag(prow, 32, hi);
#pragma unroll
    for (int ds = 0; ds < 4; ++ds) {
      const _Float16* vcol = &Vt[ds * 16 + ln][0];
      o[ds] = wmma16(ap0, load_b_frag(vcol, 0, hi), o[ds]);
      o[ds] = wmma16(ap1, load_b_frag(vcol, 32, hi), o[ds]);
    }
  }

  // normalize and write out in [B,S,H*Dk] f16
#pragma unroll
  for (int ds = 0; ds < 4; ++ds) {
#pragma unroll
    for (int r = 0; r < 8; ++r) {
      int q_g = qw + 8 * hi + r;
      size_t tok = (size_t)b * Sq + q_g;
      float ov = o[ds][r] / lr[r];
      Ah[tok * Dm + (size_t)h * Dk + ds * 16 + ln] = (_Float16)ov;
    }
  }
}

// ---------------------------------------------------------------------------
// Kernel 3: out = Ah(f16) @ Wo(f32->f16 inline), f32 result to d_out.
// A tile staged via async global->LDS.
// ---------------------------------------------------------------------------
__global__ __launch_bounds__(256) void out_proj_kernel(
    const _Float16* __restrict__ Ahp, const float* __restrict__ Wo,
    float* __restrict__ out) {
  __shared__ __align__(16) _Float16 Xt[64][72];
  __shared__ __align__(16) _Float16 Wt[32][72];

  const int tid  = threadIdx.x;
  const int wave = tid >> 5, lane = tid & 31;
  const int hi   = lane >> 4, ln = lane & 15;
  const int m0 = blockIdx.x * 64, n0 = blockIdx.y * 32;
  const int mw = (wave >> 1) * 16, nw = (wave & 1) * 16;

  v8f acc = {0.f, 0.f, 0.f, 0.f, 0.f, 0.f, 0.f, 0.f};

  for (int k0 = 0; k0 < Dm; k0 += 64) {
    __syncthreads();
    {  // A tile (f16): async DMA into LDS
      int row = tid >> 2, seg = tid & 3;
      async_copy32(Ahp + (size_t)(m0 + row) * Dm + k0 + seg * 16,
                   &Xt[row][seg * 16]);
    }
    {  // Wo tile transposed, f32 -> f16
      int k = tid >> 3, ng = tid & 7;
#pragma unroll
      for (int i = 0; i < 2; ++i) {
        int kk = k + i * 32;
        v4f f = *(const v4f*)(Wo + (size_t)(k0 + kk) * Dm + n0 + ng * 4);
        Wt[ng * 4 + 0][kk] = (_Float16)f.x;
        Wt[ng * 4 + 1][kk] = (_Float16)f.y;
        Wt[ng * 4 + 2][kk] = (_Float16)f.z;
        Wt[ng * 4 + 3][kk] = (_Float16)f.w;
      }
    }
    async_wait0();
    __syncthreads();
    const _Float16* arow = &Xt[mw + ln][0];
    const _Float16* bcol = &Wt[nw + ln][0];
#pragma unroll
    for (int kc = 0; kc < 64; kc += 32) {
      acc = wmma16(load_a_frag(arow, kc, hi), load_b_frag(bcol, kc, hi), acc);
    }
  }

  const int n_g = n0 + nw + ln;
#pragma unroll
  for (int r = 0; r < 8; ++r) {
    int m_g = m0 + mw + 8 * hi + r;
    out[(size_t)m_g * Dm + n_g] = acc[r];
  }
}

// ---------------------------------------------------------------------------
extern "C" void kernel_launch(void* const* d_in, const int* in_sizes, int n_in,
                              void* d_out, int out_size, void* d_ws,
                              size_t ws_size, hipStream_t stream) {
  const float* x  = (const float*)d_in[0];
  const float* Wq = (const float*)d_in[1];
  const float* Wk = (const float*)d_in[2];
  const float* Wv = (const float*)d_in[3];
  const float* Wo = (const float*)d_in[4];
  const int*   tp = (const int*)d_in[5];
  float* out = (float*)d_out;

  const size_t NT = (size_t)Bq * Sq;        // 4096 tokens
  _Float16* Qh = (_Float16*)d_ws;           //  8 MB
  _Float16* Kh = Qh + NT * Dm;              //  8 MB
  _Float16* Vh = Kh + NT * Dm;              //  8 MB
  _Float16* Ah = Vh + NT * Dm;              //  8 MB   (total 32 MB)

  qkv_rope_kernel<<<dim3(NT / 64, Dm / 32, 3), 256, 0, stream>>>(
      x, Wq, Wk, Wv, tp, Qh, Kh, Vh);

  attn_kernel<<<dim3(Sq / 128, Bq * Hh), 256, 0, stream>>>(Qh, Kh, Vh, Ah);

  out_proj_kernel<<<dim3(NT / 64, Dm / 32), 256, 0, stream>>>(Ah, Wo, out);
}
